// DetectionCNN_61332132986895
// MI455X (gfx1250) — matrix operand order repaired and checked by reading, not compiled, not run on hardware
//
#include <hip/hip_runtime.h>
#include <hip/hip_bf16.h>

// ---------------------------------------------------------------------------
// CDNA5 (gfx1250) implicit-GEMM detection CNN.
// bf16 activations + BN-folded bf16 weights, f32 WMMA accumulation via
// v_wmma_f32_16x16x32_bf16. wave32; 8 waves/block; 128x64 output tile
// (each wave: 16x64 strip = 4 WMMA accumulators per K-chunk).
// B tiles staged into LDS with GLOBAL_LOAD_ASYNC_TO_LDS_B128 when available.
// ---------------------------------------------------------------------------

typedef __bf16 bf16_t;
typedef __attribute__((ext_vector_type(16))) __bf16 v16bf;
typedef __attribute__((ext_vector_type(8)))  __bf16 v8bf;
typedef __attribute__((ext_vector_type(8)))  float  v8f;
typedef int v4i_t __attribute__((vector_size(4 * sizeof(int))));

#define BN_EPS 1e-5f
#define LDS_STRIDE 40   // 32 payload + 8 pad (80B rows: 16B aligned, bank-skewed)

#if defined(__AMDGCN__) && __has_builtin(__builtin_amdgcn_global_load_async_to_lds_b128)
#define HAVE_ASYNC_LDS 1
#else
#define HAVE_ASYNC_LDS 0
#endif

__device__ __forceinline__ void wait_async0() {
#if HAVE_ASYNC_LDS
#if __has_builtin(__builtin_amdgcn_s_wait_asynccnt)
  __builtin_amdgcn_s_wait_asynccnt(0);
#else
  asm volatile("s_wait_asynccnt 0" ::: "memory");
#endif
#endif
}

__device__ __forceinline__ v16bf cat8(v8bf lo, v8bf hi) {
  return __builtin_shufflevector(lo, hi, 0, 1, 2, 3, 4, 5, 6, 7,
                                 8, 9, 10, 11, 12, 13, 14, 15);
}

// Stage one 16B sliver of the B tile (weights, [Cout][Kpad] rows) into LDS.
// Thread t handles channel row c=t>>2, 16B quarter q=t&3 of the 64B k-chunk.
__device__ __forceinline__ void copy_b_tile(const bf16_t* __restrict__ Bp,
                                            bf16_t* ldsB, int n0, int Kpad,
                                            int k0, int tid) {
  int c = tid >> 2, q = tid & 3;
  const bf16_t* src = &Bp[(size_t)(n0 + c) * Kpad + k0 + 8 * q];
  bf16_t* dst = &ldsB[c * LDS_STRIDE + 8 * q];
#if HAVE_ASYNC_LDS
  // Builtin signature (from toolchain diagnostics): (AS1 v4i*, AS3 v4i*, Ii, Ii)
  __builtin_amdgcn_global_load_async_to_lds_b128(
      (__attribute__((address_space(1))) v4i_t*)src,
      (__attribute__((address_space(3))) v4i_t*)dst, 0, 0);
#else
  *(v8bf*)dst = *(const v8bf*)src;
#endif
}

// ---------------------------- utility kernels ------------------------------

__global__ void k_f32_to_bf16(const float* __restrict__ src,
                              bf16_t* __restrict__ dst, int n) {
  int i = blockIdx.x * 256 + threadIdx.x;
  if (i < n) dst[i] = (bf16_t)src[i];
}

// Pack conv weights (Cout,Cin,kh,kw) -> [Cout][Kpad] bf16, BN scale folded in.
// g==nullptr => plain conv (scale 1). Rows contiguous -> async-LDS friendly.
__global__ void k_pack_w(const float* __restrict__ w, const float* __restrict__ g,
                         const float* __restrict__ v, bf16_t* __restrict__ Bp,
                         int Cout, int K, int Kpad) {
  int i = blockIdx.x * 256 + threadIdx.x;
  if (i >= Kpad * Cout) return;
  int co = i / Kpad, kk = i - co * Kpad;
  float val = 0.f;
  if (kk < K) {
    float s = (g != nullptr) ? g[co] * rsqrtf(v[co] + BN_EPS) : 1.f;
    val = w[(size_t)co * K + kk] * s;
  }
  Bp[i] = (bf16_t)val;
}

// Pack ConvTranspose2d(k=4,s=2,p=1) weights (Cin=256,Cout=128,4,4) into four
// per-output-parity GEMM-B matrices [4][128][1024], BN scale folded.
__global__ void k_pack_w_deconv(const float* __restrict__ w,
                                const float* __restrict__ g,
                                const float* __restrict__ v,
                                bf16_t* __restrict__ Bp) {
  int i = blockIdx.x * 256 + threadIdx.x;
  if (i >= 4 * 128 * 1024) return;
  int par = i >> 17;             // 128*1024 = 2^17
  int rem = i & (131072 - 1);
  int co  = rem >> 10;
  int kk  = rem & 1023;          // K index: ci*4 + ty*2 + tx
  int ry = par >> 1, rx = par & 1;          // output pixel parity (oy&1, ox&1)
  int ci = kk >> 2, ty = (kk >> 1) & 1, tx = kk & 1;
  int ky = (1 - ry) + 2 * ty;               // kernel tap for this parity
  int kx = (1 - rx) + 2 * tx;
  float s = g[co] * rsqrtf(v[co] + BN_EPS);
  float val = w[(((size_t)ci * 128 + co) * 4 + ky) * 4 + kx] * s;
  Bp[i] = (bf16_t)val;
}

// bias' = (b - m) * g/sqrt(v+eps) + beta   (or plain b when g==nullptr)
__global__ void k_make_bias(const float* __restrict__ b, const float* __restrict__ g,
                            const float* __restrict__ beta, const float* __restrict__ m,
                            const float* __restrict__ v, float* __restrict__ bias, int C) {
  int i = blockIdx.x * 256 + threadIdx.x;
  if (i >= C) return;
  if (g != nullptr) {
    float s = g[i] * rsqrtf(v[i] + BN_EPS);
    bias[i] = (b[i] - m[i]) * s + beta[i];
  } else {
    bias[i] = b[i];
  }
}

__global__ void k_maxpool2(const bf16_t* __restrict__ in, bf16_t* __restrict__ out,
                           int BC, int H, int W) {
  int Ho = H >> 1, Wo = W >> 1;
  int i = blockIdx.x * 256 + threadIdx.x;
  if (i >= BC * Ho * Wo) return;
  int x = i % Wo, t = i / Wo;
  int y = t % Ho, bc = t / Ho;
  const bf16_t* p = in + ((size_t)bc * H + 2 * y) * W + 2 * x;
  float a = (float)p[0], b = (float)p[1], c = (float)p[W], d = (float)p[W + 1];
  out[i] = (bf16_t)fmaxf(fmaxf(a, b), fmaxf(c, d));
}

// ---------------------- WMMA implicit-GEMM convolution ---------------------
// in:  bf16 NCHW (Bn,Cin,H,W)   Bp: bf16 [Cout][Kpad] (BN-folded)
// out: bf16 NCHW (Bn,Cout,H,W)  stride 1, square kernel kh x kw, pad.
// Block: 128 pixels x 64 channels; wave w owns pixel strip [16w,16w+16).
__global__ void __launch_bounds__(256)
k_conv_wmma(const bf16_t* __restrict__ in, const bf16_t* __restrict__ Bp,
            const float* __restrict__ bias, bf16_t* __restrict__ out,
            int Bn, int Cin, int H, int W, int Cout,
            int kh, int kw, int pad, int Kpad, int relu) {
  __shared__ bf16_t ldsA[128 * LDS_STRIDE];  // [pixel][k]
  __shared__ bf16_t ldsB[64 * LDS_STRIDE];   // [chan][k]
  const int tid  = threadIdx.x;
  const int wave = tid >> 5, lane = tid & 31;
  const int half = lane >> 4, l15 = lane & 15;
  const int HW = H * W;
  const int m0 = blockIdx.x * 128;          // pixel tile base (b*H*W + y*W + x)
  const int n0 = blockIdx.y * 64;           // channel tile base
  const int khw = kh * kw;
  const int K   = Cin * khw;
  const int mw  = wave << 4;                // wave M strip
  // Per-thread A-gather geometry is K-invariant: fixed kq column, 16 fixed
  // pixels (p = tid>>5 + 8i). Precompute decomposition once.
  const int kq = tid & 31, p0 = tid >> 5;
  int yy[16], xx[16], gbase[16];
#pragma unroll
  for (int i = 0; i < 16; ++i) {
    int pix = m0 + p0 + 8 * i;
    int b = pix / HW, s = pix - b * HW;
    int y = s / W, x = s - y * W;
    yy[i] = y; xx[i] = x;
    gbase[i] = b * Cin * HW + y * W + x;
  }
  v8f acc[4];
#pragma unroll
  for (int t = 0; t < 4; ++t) acc[t] = (v8f){0.f,0.f,0.f,0.f,0.f,0.f,0.f,0.f};

  for (int k0 = 0; k0 < Kpad; k0 += 32) {
    copy_b_tile(Bp, ldsB, n0, Kpad, k0, tid);     // async DMA -> LDS
    // A tile: one (ci,ky,kx) decomposition per chunk, then 16 cheap gathers.
    int kk = k0 + kq;
    bool kvalid = kk < K;
    int dy = 0, dx = 0, koff = 0;
    if (kvalid) {
      int ci = kk / khw, r = kk - ci * khw;
      int ky = r / kw, kx2 = r - ky * kw;
      dy = ky - pad; dx = kx2 - pad;
      koff = ci * HW + dy * W + dx;
    }
#pragma unroll
    for (int i = 0; i < 16; ++i) {
      float val = 0.f;
      int iy = yy[i] + dy, ix = xx[i] + dx;
      if (kvalid && (unsigned)iy < (unsigned)H && (unsigned)ix < (unsigned)W)
        val = (float)in[gbase[i] + koff];
      ldsA[(p0 + 8 * i) * LDS_STRIDE + kq] = (bf16_t)val;
    }
    wait_async0();
    __syncthreads();
    // A fragment: lane(h,m) holds K in [8h,8h+8) and [16+8h,16+8h+8)
    v8bf alo = *(const v8bf*)&ldsA[(mw + l15) * LDS_STRIDE + 8 * half];
    v8bf ahi = *(const v8bf*)&ldsA[(mw + l15) * LDS_STRIDE + 16 + 8 * half];
    v16bf af = cat8(alo, ahi);
    // 4 B fragments: lane(h,n) holds K in [8h,8h+8) and [16+8h,16+8h+8)
#pragma unroll
    for (int t = 0; t < 4; ++t) {
      const bf16_t* pb = &ldsB[(16 * t + l15) * LDS_STRIDE + 8 * half];
      v16bf bfr = cat8(*(const v8bf*)pb, *(const v8bf*)(pb + 16));
      acc[t] = __builtin_amdgcn_wmma_f32_16x16x32_bf16(false, af, false, bfr,
                                                       (short)0, acc[t],
                                                       false, false);
    }
    __syncthreads();
  }

  // Epilogue: lane's 8 acc values = 8 consecutive pixels of one channel.
  int pix0 = m0 + mw + 8 * half;
  int b = pix0 / HW, s = pix0 - b * HW;
#pragma unroll
  for (int t = 0; t < 4; ++t) {
    int c = n0 + 16 * t + l15;
    float bb = bias[c];
    v8bf o;
#pragma unroll
    for (int r = 0; r < 8; ++r) {
      float f = acc[t][r] + bb;
      if (relu) f = fmaxf(f, 0.f);
      o[r] = (bf16_t)f;
    }
    *(v8bf*)&out[((size_t)b * Cout + c) * HW + s] = o;  // 16B packed store
  }
}

// --------------- WMMA ConvTranspose2d(k=4,s=2,p=1) via parity --------------
// in (Bn,256,192,192) -> out (Bn,128,384,384); Bp4: [4][128][1024]
// blockIdx.z = output parity (ry,rx); per-parity it is a 2x2-tap dense conv.
__global__ void __launch_bounds__(256)
k_deconv_wmma(const bf16_t* __restrict__ in, const bf16_t* __restrict__ Bp4,
              const float* __restrict__ bias, bf16_t* __restrict__ out, int Bn) {
  const int Hi = 192, Wi = 192, Cin = 256, Cout = 128, Ho = 384, Wo = 384;
  const int Kpad = 1024;
  __shared__ bf16_t ldsA[128 * LDS_STRIDE];
  __shared__ bf16_t ldsB[64 * LDS_STRIDE];
  const int tid  = threadIdx.x;
  const int wave = tid >> 5, lane = tid & 31;
  const int half = lane >> 4, l15 = lane & 15;
  const int par = blockIdx.z;
  const int ry = par >> 1, rx = par & 1;
  const bf16_t* Bp = Bp4 + (size_t)par * Cout * Kpad;
  const int HWq = Hi * Wi;                  // per-parity output grid = 192x192
  const int m0 = blockIdx.x * 128;
  const int n0 = blockIdx.y * 64;
  const int mw = wave << 4;
  const int kq = tid & 31, p0 = tid >> 5;
  int jY[16], jX[16], gbase[16];
#pragma unroll
  for (int i = 0; i < 16; ++i) {
    int pix = m0 + p0 + 8 * i;
    int b = pix / HWq, s = pix - b * HWq;
    int jy = s / Wi, jx = s - jy * Wi;      // out pixel (2*jy+ry, 2*jx+rx)
    jY[i] = jy + ry; jX[i] = jx + rx;
    gbase[i] = b * Cin * HWq + (jy + ry) * Wi + (jx + rx);
  }
  v8f acc[4];
#pragma unroll
  for (int t = 0; t < 4; ++t) acc[t] = (v8f){0.f,0.f,0.f,0.f,0.f,0.f,0.f,0.f};

  for (int k0 = 0; k0 < Kpad; k0 += 32) {
    copy_b_tile(Bp, ldsB, n0, Kpad, k0, tid);
    int kk = k0 + kq;
    int ci = kk >> 2, ty = (kk >> 1) & 1, tx = kk & 1;
    int koff = ci * HWq - ty * Wi - tx;
#pragma unroll
    for (int i = 0; i < 16; ++i) {
      float val = 0.f;
      int iy = jY[i] - ty, ix = jX[i] - tx;
      if ((unsigned)iy < (unsigned)Hi && (unsigned)ix < (unsigned)Wi)
        val = (float)in[gbase[i] + koff];
      ldsA[(p0 + 8 * i) * LDS_STRIDE + kq] = (bf16_t)val;
    }
    wait_async0();
    __syncthreads();
    v8bf alo = *(const v8bf*)&ldsA[(mw + l15) * LDS_STRIDE + 8 * half];
    v8bf ahi = *(const v8bf*)&ldsA[(mw + l15) * LDS_STRIDE + 16 + 8 * half];
    v16bf af = cat8(alo, ahi);
#pragma unroll
    for (int t = 0; t < 4; ++t) {
      const bf16_t* pb = &ldsB[(16 * t + l15) * LDS_STRIDE + 8 * half];
      v16bf bfr = cat8(*(const v8bf*)pb, *(const v8bf*)(pb + 16));
      acc[t] = __builtin_amdgcn_wmma_f32_16x16x32_bf16(false, af, false, bfr,
                                                       (short)0, acc[t],
                                                       false, false);
    }
    __syncthreads();
  }

  int pix0 = m0 + mw + 8 * half;
  int b = pix0 / HWq, s = pix0 - b * HWq;
  int jy0 = s / Wi, jx0 = s - jy0 * Wi;
  int oy = 2 * jy0 + ry;
#pragma unroll
  for (int t = 0; t < 4; ++t) {
    int c = n0 + 16 * t + l15;
    float bb = bias[c];
    size_t rowbase = (((size_t)b * Cout + c) * Ho + oy) * Wo;
#pragma unroll
    for (int r = 0; r < 8; ++r) {
      float f = fmaxf(acc[t][r] + bb, 0.f);  // up1 is always BN+ReLU
      out[rowbase + (2 * (jx0 + r) + rx)] = (bf16_t)f;   // stride-2 scatter
    }
  }
}

// ---------------- 1x1 head convs (Cout 1/2) + activation, fp32 out ---------
__global__ void k_head1x1(const bf16_t* __restrict__ in, const float* __restrict__ w,
                          const float* __restrict__ bias, float* __restrict__ out,
                          int Bn, int HW, int Co, int act) {
  int i = blockIdx.x * 256 + threadIdx.x;
  if (i >= Bn * Co * HW) return;
  int s = i % HW, t = i / HW;
  int co = t % Co, b = t / Co;
  const bf16_t* ip = in + (size_t)b * 64 * HW + s;
  float acc = bias[co];
#pragma unroll 8
  for (int c = 0; c < 64; ++c) acc += (float)ip[(size_t)c * HW] * w[co * 64 + c];
  if (act == 1) acc = 1.f / (1.f + __expf(-acc));
  out[i] = acc;
}

// ------------------------------- peak decode -------------------------------
__global__ void k_decode(const float* __restrict__ hm, const float* __restrict__ off,
                         const float* __restrict__ sz, float* __restrict__ out,
                         int Bn, int Hh, int Wd) {
  int HW = Hh * Wd;
  int i = blockIdx.x * 256 + threadIdx.x;
  if (i >= Bn * HW) return;
  int x = i % Wd, t = i / Wd;
  int y = t % Hh, b = t / Hh;
  const float* hb = hm + (size_t)b * HW;
  float v = hb[y * Wd + x];
  float mx = -INFINITY;
  for (int dy = -1; dy <= 1; ++dy)
    for (int dx = -1; dx <= 1; ++dx) {
      int yy = y + dy, xx = x + dx;
      if ((unsigned)yy < (unsigned)Hh && (unsigned)xx < (unsigned)Wd)
        mx = fmaxf(mx, hb[yy * Wd + xx]);
    }
  bool mask = (v == mx) && (v > 0.f);
  const float* ob = off + (size_t)b * 2 * HW;
  float xf = (float)x + ob[y * Wd + x];
  float yf = (float)y + ob[HW + y * Wd + x];
  int ix = (int)fminf(fmaxf(xf, 0.f), (float)(Wd - 1));
  int iy = (int)fminf(fmaxf(yf, 0.f), (float)(Hh - 1));
  int flat = iy * Wd + ix;
  const float* sb = sz + (size_t)b * 2 * HW;
  float szw = sb[flat] * (float)Wd;
  float szh = sb[HW + flat] * (float)Hh;
  float x1 = fminf(fmaxf(xf - 0.5f * szw, 0.f), (float)(Wd - 1));
  float y1 = fminf(fmaxf(yf - 0.5f * szh, 0.f), (float)(Hh - 1));
  float x2 = fminf(fmaxf(xf + 0.5f * szw, 0.f), (float)(Wd - 1));
  float y2 = fminf(fmaxf(yf + 0.5f * szh, 0.f), (float)(Hh - 1));
  bool valid = mask && (x2 > x1) && (y2 > y1);
  float* o = out + (size_t)i * 5;
  o[0] = valid ? v : 0.f;
  o[1] = valid ? x1 : 0.f;
  o[2] = valid ? y1 : 0.f;
  o[3] = valid ? x2 : 0.f;
  o[4] = valid ? y2 : 0.f;
}

// ------------------------------- launcher ----------------------------------
extern "C" void kernel_launch(void* const* d_in, const int* in_sizes, int n_in,
                              void* d_out, int out_size, void* d_ws, size_t ws_size,
                              hipStream_t stream) {
  (void)in_sizes; (void)n_in; (void)out_size; (void)ws_size;
  // setup_inputs() dict order: x, cb1{w,b,g,beta,m,v}, cb2{...}, cb3{...},
  // up1{w,b,g,beta,m,v}, up2{...}, hm1w,hm1b,hm2w,hm2b, of1w..of2b, sz1w..sz2b
  const float* x = (const float*)d_in[0];
  const float *cb1w=(const float*)d_in[1],  *cb1b=(const float*)d_in[2],
              *cb1g=(const float*)d_in[3],  *cb1be=(const float*)d_in[4],
              *cb1m=(const float*)d_in[5],  *cb1v=(const float*)d_in[6];
  const float *cb2w=(const float*)d_in[7],  *cb2b=(const float*)d_in[8],
              *cb2g=(const float*)d_in[9],  *cb2be=(const float*)d_in[10],
              *cb2m=(const float*)d_in[11], *cb2v=(const float*)d_in[12];
  const float *cb3w=(const float*)d_in[13], *cb3b=(const float*)d_in[14],
              *cb3g=(const float*)d_in[15], *cb3be=(const float*)d_in[16],
              *cb3m=(const float*)d_in[17], *cb3v=(const float*)d_in[18];
  const float *u1w=(const float*)d_in[19],  *u1b=(const float*)d_in[20],
              *u1g=(const float*)d_in[21],  *u1be=(const float*)d_in[22],
              *u1m=(const float*)d_in[23],  *u1v=(const float*)d_in[24];
  const float *u2w=(const float*)d_in[25],  *u2b=(const float*)d_in[26],
              *u2g=(const float*)d_in[27],  *u2be=(const float*)d_in[28],
              *u2m=(const float*)d_in[29],  *u2v=(const float*)d_in[30];
  const float *hm1w=(const float*)d_in[31], *hm1b=(const float*)d_in[32],
              *hm2w=(const float*)d_in[33], *hm2b=(const float*)d_in[34];
  const float *of1w=(const float*)d_in[35], *of1b=(const float*)d_in[36],
              *of2w=(const float*)d_in[37], *of2b=(const float*)d_in[38];
  const float *sz1w=(const float*)d_in[39], *sz1b=(const float*)d_in[40],
              *sz2w=(const float*)d_in[41], *sz2b=(const float*)d_in[42];

  const int B = 8, H = 384, W = 384, HW = H * W;
  const int Hp = 192, Wp = 192, HWp = Hp * Wp;

  char* ws = (char*)d_ws;
  size_t off = 0;
  auto alloc = [&](size_t bytes) -> char* {
    char* p = ws + off;
    off += (bytes + 255) & ~((size_t)255);
    return p;
  };
  bf16_t* xbf   = (bf16_t*)alloc((size_t)B * HW * 2);           // input bf16
  bf16_t* actA  = (bf16_t*)alloc((size_t)B * 64 * HW * 2);      // 151 MB
  bf16_t* actB  = (bf16_t*)alloc((size_t)B * 64 * HW * 2);      // 151 MB
  bf16_t* poolb = (bf16_t*)alloc((size_t)B * 64 * HWp * 2);     // 38 MB
  bf16_t* up1o  = (bf16_t*)alloc((size_t)B * 128 * HW * 2);     // 302 MB
  // cb3 out (8,256,192,192) bf16 == actA size -> reuse actA after cb2.
  bf16_t* cb3o  = actA;
  float* hmb  = (float*)alloc((size_t)B * 1 * HW * 4);
  float* offb = (float*)alloc((size_t)B * 2 * HW * 4);
  float* szb  = (float*)alloc((size_t)B * 2 * HW * 4);
  bf16_t* pw1  = (bf16_t*)alloc((size_t)64 * 96 * 2);
  bf16_t* pw2  = (bf16_t*)alloc((size_t)64 * 1600 * 2);
  bf16_t* pw3  = (bf16_t*)alloc((size_t)256 * 64 * 2);
  bf16_t* pwu1 = (bf16_t*)alloc((size_t)4 * 128 * 1024 * 2);
  bf16_t* pwu2 = (bf16_t*)alloc((size_t)64 * 1152 * 2);
  bf16_t* pwh1 = (bf16_t*)alloc((size_t)64 * 1600 * 2);
  bf16_t* pwo1 = (bf16_t*)alloc((size_t)64 * 576 * 2);
  bf16_t* pws1 = (bf16_t*)alloc((size_t)64 * 576 * 2);
  float* pb1  = (float*)alloc(64 * 4);
  float* pb2  = (float*)alloc(64 * 4);
  float* pb3  = (float*)alloc(256 * 4);
  float* pbu1 = (float*)alloc(128 * 4);
  float* pbu2 = (float*)alloc(64 * 4);
  float* pbh1 = (float*)alloc(64 * 4);
  float* pbo1 = (float*)alloc(64 * 4);
  float* pbs1 = (float*)alloc(64 * 4);

  dim3 blk(256);
  auto cdiv = [](int a, int b) { return (a + b - 1) / b; };

  // ---- prep: dtype convert + weight packing (BN folded) ----
  k_f32_to_bf16<<<cdiv(B * HW, 256), blk, 0, stream>>>(x, xbf, B * HW);
  k_pack_w<<<cdiv(96 * 64, 256),   blk, 0, stream>>>(cb1w, cb1g, cb1v, pw1, 64, 81, 96);
  k_pack_w<<<cdiv(1600 * 64, 256), blk, 0, stream>>>(cb2w, cb2g, cb2v, pw2, 64, 1600, 1600);
  k_pack_w<<<cdiv(64 * 256, 256),  blk, 0, stream>>>(cb3w, cb3g, cb3v, pw3, 256, 64, 64);
  k_pack_w_deconv<<<cdiv(4 * 128 * 1024, 256), blk, 0, stream>>>(u1w, u1g, u1v, pwu1);
  k_pack_w<<<cdiv(1152 * 64, 256), blk, 0, stream>>>(u2w, u2g, u2v, pwu2, 64, 1152, 1152);
  k_pack_w<<<cdiv(1600 * 64, 256), blk, 0, stream>>>(hm1w, nullptr, nullptr, pwh1, 64, 1600, 1600);
  k_pack_w<<<cdiv(576 * 64, 256),  blk, 0, stream>>>(of1w, nullptr, nullptr, pwo1, 64, 576, 576);
  k_pack_w<<<cdiv(576 * 64, 256),  blk, 0, stream>>>(sz1w, nullptr, nullptr, pws1, 64, 576, 576);
  k_make_bias<<<1, blk, 0, stream>>>(cb1b, cb1g, cb1be, cb1m, cb1v, pb1, 64);
  k_make_bias<<<1, blk, 0, stream>>>(cb2b, cb2g, cb2be, cb2m, cb2v, pb2, 64);
  k_make_bias<<<1, blk, 0, stream>>>(cb3b, cb3g, cb3be, cb3m, cb3v, pb3, 256);
  k_make_bias<<<1, blk, 0, stream>>>(u1b,  u1g,  u1be,  u1m,  u1v,  pbu1, 128);
  k_make_bias<<<1, blk, 0, stream>>>(u2b,  u2g,  u2be,  u2m,  u2v,  pbu2, 64);
  k_make_bias<<<1, blk, 0, stream>>>(hm1b, nullptr, nullptr, nullptr, nullptr, pbh1, 64);
  k_make_bias<<<1, blk, 0, stream>>>(of1b, nullptr, nullptr, nullptr, nullptr, pbo1, 64);
  k_make_bias<<<1, blk, 0, stream>>>(sz1b, nullptr, nullptr, nullptr, nullptr, pbs1, 64);

  // ---- backbone (all WMMA) ----
  dim3 gFull(B * HW / 128, 1, 1);    // 9216 pixel tiles, Cout=64
  k_conv_wmma<<<gFull, blk, 0, stream>>>(xbf, pw1, pb1, actA,
                                         B, 1, H, W, 64, 9, 9, 4, 96, 1);      // cb1
  k_conv_wmma<<<gFull, blk, 0, stream>>>(actA, pw2, pb2, actB,
                                         B, 64, H, W, 64, 5, 5, 2, 1600, 1);   // cb2
  k_maxpool2<<<cdiv(B * 64 * HWp, 256), blk, 0, stream>>>(actB, poolb, B * 64, H, W);
  dim3 g3(B * HWp / 128, 4, 1);      // Cout=256
  k_conv_wmma<<<g3, blk, 0, stream>>>(poolb, pw3, pb3, cb3o,
                                      B, 64, Hp, Wp, 256, 1, 1, 0, 64, 1);     // cb3
  dim3 gd(B * HWp / 128, 2, 4);      // Cout=128, 4 output parities
  k_deconv_wmma<<<gd, blk, 0, stream>>>(cb3o, pwu1, pbu1, up1o, B);            // up1
  k_conv_wmma<<<gFull, blk, 0, stream>>>(up1o, pwu2, pbu2, actB,
                                         B, 128, H, W, 64, 3, 3, 1, 1152, 1);  // up2

  // ---- heads (mid convs WMMA, 1x1 tails scalar) ----
  k_conv_wmma<<<gFull, blk, 0, stream>>>(actB, pwh1, pbh1, actA,
                                         B, 64, H, W, 64, 5, 5, 2, 1600, 1);   // hm1
  k_head1x1<<<cdiv(B * 1 * HW, 256), blk, 0, stream>>>(actA, hm2w, hm2b, hmb,
                                                       B, HW, 1, 1);           // sigmoid
  k_conv_wmma<<<gFull, blk, 0, stream>>>(actB, pwo1, pbo1, actA,
                                         B, 64, H, W, 64, 3, 3, 1, 576, 1);    // of1
  k_head1x1<<<cdiv(B * 2 * HW, 256), blk, 0, stream>>>(actA, of2w, of2b, offb,
                                                       B, HW, 2, 0);           // linear
  k_conv_wmma<<<gFull, blk, 0, stream>>>(actB, pws1, pbs1, actA,
                                         B, 64, H, W, 64, 3, 3, 1, 576, 1);    // sz1
  k_head1x1<<<cdiv(B * 2 * HW, 256), blk, 0, stream>>>(actA, sz2w, sz2b, szb,
                                                       B, HW, 2, 1);           // sigmoid

  // ---- decode to dense [B,384,384,5] ----
  k_decode<<<cdiv(B * HW, 256), blk, 0, stream>>>(hmb, offb, szb,
                                                  (float*)d_out, B, H, W);
}
